// FusionNet_46557445489144
// MI455X (gfx1250) — compile-verified, matrix operand order checked by reading, not run on hardware
//
#include <hip/hip_runtime.h>
#include <hip/hip_bf16.h>

// ---------------------------------------------------------------------------
// Types for CDNA5 WMMA (wave32): 16x16x32 bf16 -> f32 accumulate
// ---------------------------------------------------------------------------
typedef __bf16 bf16_t;
typedef bf16_t v16bf __attribute__((ext_vector_type(16)));
typedef float  v8f   __attribute__((ext_vector_type(8)));

#define EPSN 1e-12f
#define NEGBIG -3.0e38f

// ---------------------------------------------------------------------------
// feat_feature = concat(vertices, xyz, normal) -> (B*N, 9)
// ---------------------------------------------------------------------------
__global__ void build_featF(const float* __restrict__ v, const float* __restrict__ x,
                            const float* __restrict__ nm, float* __restrict__ out, int BN) {
    int t = blockIdx.x * blockDim.x + threadIdx.x;
    if (t >= BN * 9) return;
    int d = t % 9, r = t / 9;
    const float* s = (d < 3) ? v : ((d < 6) ? x : nm);
    out[t] = s[(size_t)r * 3 + (d % 3)];
}

// ---------------------------------------------------------------------------
// KNN: for each of B*Nq query points, k nearest among Nsrc points (self excluded)
// src rows: D floats, batch stride bstride (in floats).
// ---------------------------------------------------------------------------
__global__ void knn_kernel(const float* __restrict__ src, int* __restrict__ out,
                           int B, int Nsrc, int Nq, int D, int bstride, int k) {
    int t = blockIdx.x * blockDim.x + threadIdx.x;
    if (t >= B * Nq) return;
    int b = t / Nq, n = t % Nq;
    const float* base = src + (size_t)b * bstride;
    float q[9];
    for (int d = 0; d < D; ++d) q[d] = base[(size_t)n * D + d];
    float bestd[10];
    int   besti[10];
    for (int j = 0; j < k; ++j) { bestd[j] = 3.4e38f; besti[j] = 0; }
    for (int m = 0; m < Nsrc; ++m) {
        if (m == n) continue;
        float d2 = 0.f;
        for (int d = 0; d < D; ++d) { float df = base[(size_t)m * D + d] - q[d]; d2 += df * df; }
        if (d2 < bestd[k - 1]) {
            int j = k - 1;
            while (j > 0 && bestd[j - 1] > d2) { bestd[j] = bestd[j - 1]; besti[j] = besti[j - 1]; --j; }
            bestd[j] = d2; besti[j] = m;
        }
    }
    for (int j = 0; j < k; ++j) out[(size_t)t * k + j] = besti[j];
}

// ---------------------------------------------------------------------------
// Normalized neighbor directions: dirn[b,n,k,:] = normalize(v[idx]-v[n])
// ---------------------------------------------------------------------------
__global__ void dirn_kernel(const float* __restrict__ v, const int* __restrict__ idx,
                            float* __restrict__ out, int B, int Nq, int K, int D, int bstride) {
    int t = blockIdx.x * blockDim.x + threadIdx.x;
    if (t >= B * Nq * K) return;
    int k = t % K;
    int n = (t / K) % Nq;
    int b = t / (K * Nq);
    const float* base = v + (size_t)b * bstride;
    int j = idx[((size_t)b * Nq + n) * K + k];
    float dv[9]; float s = 0.f;
    for (int d = 0; d < D; ++d) {
        float df = base[(size_t)j * D + d] - base[(size_t)n * D + d];
        dv[d] = df; s += df * df;
    }
    float inv = 1.0f / fmaxf(sqrtf(s), EPSN);
    for (int d = 0; d < D; ++d) out[(size_t)t * D + d] = dv[d] * inv;
}

// normalize support directions per column: dirs (D,S) -> out (D,S)
__global__ void norm_cols(const float* __restrict__ dirs, float* __restrict__ out, int D, int S) {
    int s = blockIdx.x * blockDim.x + threadIdx.x;
    if (s >= S) return;
    float sum = 0.f;
    for (int d = 0; d < D; ++d) { float v = dirs[(size_t)d * S + s]; sum += v * v; }
    float inv = 1.0f / fmaxf(sqrtf(sum), EPSN);
    for (int d = 0; d < D; ++d) out[(size_t)d * S + s] = dirs[(size_t)d * S + s] * inv;
}

// ---------------------------------------------------------------------------
// conv_surface: out[bn,c] = sum_s max_k relu(dirn[bn,k] . supn[:, s*128+c])
// grid: B*Nq blocks, block = 128 threads, D=3, K=10
// ---------------------------------------------------------------------------
__global__ void conv_surface_kernel(const float* __restrict__ dirn, const float* __restrict__ supn,
                                    float* __restrict__ out, int K) {
    __shared__ float sd[30];
    int bn = blockIdx.x;
    int c = threadIdx.x;
    if (c < K * 3) sd[c] = dirn[(size_t)bn * K * 3 + c];
    __syncthreads();
    const int CK = 128, S = 7 * 128;
    float acc = 0.f;
    for (int s = 0; s < 7; ++s) {
        int col = s * CK + c;
        float sx = supn[col], sy = supn[S + col], sz = supn[2 * S + col];
        float m = 0.f; // relu(dot) >= 0
        for (int k = 0; k < K; ++k) {
            float dot = sd[k * 3] * sx + sd[k * 3 + 1] * sy + sd[k * 3 + 2] * sz;
            m = fmaxf(m, dot);
        }
        acc += m;
    }
    out[(size_t)bn * CK + c] = acc;
}

// ---------------------------------------------------------------------------
// conv_layer combine: out[bn,c] = f[bn,c] + sum_s max_k relu(dirn.sup)*f[nb, outC+s*outC+c]
// f: (B*Nq, cols) with bias already added. grid: B*Nq blocks, block = outC threads
// ---------------------------------------------------------------------------
__global__ void conv_layer_kernel(const float* __restrict__ dirn, const float* __restrict__ supn,
                                  const int* __restrict__ idx, const float* __restrict__ f,
                                  float* __restrict__ out, int Nq, int K, int D, int outC,
                                  int cols, int ostride) {
    __shared__ float sd[90];
    __shared__ int   si[10];
    int bn = blockIdx.x;
    int c = threadIdx.x;
    if (c < K * D) sd[c] = dirn[(size_t)bn * K * D + c];
    if (c < K)     si[c] = idx[(size_t)bn * K + c];
    __syncthreads();
    int b = bn / Nq;
    int S = 7 * outC;
    float acc = f[(size_t)bn * cols + c]; // center (bias included)
    for (int s = 0; s < 7; ++s) {
        int scol = s * outC + c;
        int fcol = outC + scol;
        float m = NEGBIG;
        for (int k = 0; k < K; ++k) {
            float dot = 0.f;
            for (int d = 0; d < D; ++d) dot += sd[k * D + d] * supn[(size_t)d * S + scol];
            float th = fmaxf(dot, 0.f);
            float sup = f[((size_t)b * Nq + si[k]) * cols + fcol];
            m = fmaxf(m, th * sup);
        }
        acc += m;
    }
    out[(size_t)bn * ostride + c] = acc;
}

// ---------------------------------------------------------------------------
// GEMM via WMMA with async-LDS-staged B strip.
// C[M,N] = A[M,K](bf16) * Bt[N,K](bf16, pre-transposed) + bias[N]
// Block = 256 threads = 8 waves; block covers a 128x16 C macro-tile:
//   - B strip (16 cols x K) staged once into LDS via global_load_async_to_lds_b128
//     (ASYNCcnt), column stride padded by 8 bf16 to spread LDS banks.
//   - each wave computes one 16x16 tile; A rows streamed from global (b128 pairs),
//     B fragments read from LDS (ds_load_b128 pairs).
// A lane layout : row = tm*16 + (lane&15), kslice = (lane>>4)*16, 16 contiguous bf16
// B lane layout : col = (lane&15),          kslice = (lane>>4)*16, 16 contiguous bf16
// C lane layout : C[tm*16 + r + (lane>>4)*8][tn*16 + (lane&15)] = acc[r]
// ---------------------------------------------------------------------------
__global__ void wmma_gemm_bias(const bf16_t* __restrict__ A, const bf16_t* __restrict__ Bt,
                               const float* __restrict__ bias, float* __restrict__ C,
                               int M, int N, int K) {
    extern __shared__ bf16_t sB[]; // 16 * (K+8) bf16
    const int PAD = 8;
    int tilesN = N >> 4;
    int tn = blockIdx.x % tilesN;
    int tmBase = (blockIdx.x / tilesN) * 8;

    // ---- stage B strip (16 columns of Bt, each K bf16) into LDS (async) ----
    {
        unsigned ldsBase = (unsigned)(size_t)(void*)sB;
        int chunksPerCol = K >> 3;              // 16B chunks per column
        int totalChunks = 16 * chunksPerCol;
        for (int i = threadIdx.x; i < totalChunks; i += blockDim.x) {
            int col = i / chunksPerCol;
            int cofs = (i - col * chunksPerCol) << 4;   // byte offset within column
            unsigned long long g =
                (unsigned long long)(size_t)((const char*)(Bt + ((size_t)(tn * 16 + col)) * K) + cofs);
            unsigned l = ldsBase + (unsigned)(col * (K + PAD) * 2 + cofs);
            asm volatile("global_load_async_to_lds_b128 %0, %1, off"
                         :: "v"(l), "v"(g) : "memory");
        }
        asm volatile("s_wait_asynccnt 0x0" ::: "memory");
        __syncthreads();
    }

    int lane = threadIdx.x & 31;
    int wave = threadIdx.x >> 5;
    int tm = tmBase + wave;
    int rowA = tm * 16 + (lane & 15);
    int koff = (lane >> 4) * 16;
    const bf16_t* sBcol = sB + (size_t)(lane & 15) * (K + PAD);
    v8f acc = {0.f, 0.f, 0.f, 0.f, 0.f, 0.f, 0.f, 0.f};
    for (int kk = 0; kk < K; kk += 32) {
        v16bf a = *(const v16bf*)(A + (size_t)rowA * K + kk + koff);
        v16bf b = *(const v16bf*)(sBcol + kk + koff);
        acc = __builtin_amdgcn_wmma_f32_16x16x32_bf16(
            /*neg_a=*/false, a, /*neg_b=*/false, b,
            /*c_mod=*/(short)0, acc, /*reuse_a=*/false, /*reuse_b=*/false);
    }
    int row0 = tm * 16 + (lane >> 4) * 8;
    int col  = tn * 16 + (lane & 15);
    float bv = bias[col];
#pragma unroll
    for (int r = 0; r < 8; ++r)
        C[(size_t)(row0 + r) * N + col] = acc[r] + bv;
}

__global__ void cvt_bf16(const float* __restrict__ in, bf16_t* __restrict__ out, int n) {
    int t = blockIdx.x * blockDim.x + threadIdx.x;
    if (t < n) out[t] = (bf16_t)in[t];
}

// W: (K, N) f32 -> Wt: (N, K) bf16
__global__ void transpose_cvt(const float* __restrict__ W, bf16_t* __restrict__ Wt, int K, int N) {
    int t = blockIdx.x * blockDim.x + threadIdx.x;
    if (t >= K * N) return;
    int k = t / N, n = t % N;
    Wt[(size_t)n * K + k] = (bf16_t)W[t];
}

// ---------------------------------------------------------------------------
// BatchNorm (population stats over all rows) + ReLU, in place.
// ---------------------------------------------------------------------------
__global__ void bn_stats(const float* __restrict__ x, float* __restrict__ mean,
                         float* __restrict__ var, int rows, int ld) {
    __shared__ float sh[256], sh2[256];
    int c = blockIdx.x;
    float s = 0.f, s2 = 0.f;
    for (int r = threadIdx.x; r < rows; r += blockDim.x) {
        float v = x[(size_t)r * ld + c];
        s += v; s2 += v * v;
    }
    sh[threadIdx.x] = s; sh2[threadIdx.x] = s2;
    __syncthreads();
    for (int o = blockDim.x >> 1; o > 0; o >>= 1) {
        if ((int)threadIdx.x < o) { sh[threadIdx.x] += sh[threadIdx.x + o]; sh2[threadIdx.x] += sh2[threadIdx.x + o]; }
        __syncthreads();
    }
    if (threadIdx.x == 0) {
        float m = sh[0] / rows;
        mean[c] = m;
        var[c] = sh2[0] / rows - m * m;
    }
}

__global__ void bn_apply_relu(float* __restrict__ x, const float* __restrict__ mean,
                              const float* __restrict__ var, const float* __restrict__ g,
                              const float* __restrict__ be, int rows, int ld, int C) {
    int t = blockIdx.x * blockDim.x + threadIdx.x;
    if (t >= rows * C) return;
    int r = t / C, c = t % C;
    float v = x[(size_t)r * ld + c];
    v = (v - mean[c]) * rsqrtf(var[c] + 1e-5f) * g[c] + be[c];
    x[(size_t)r * ld + c] = fmaxf(v, 0.f);
}

// ---------------------------------------------------------------------------
// pool: out[b,i,c] = max_j feat[b, idx[b,i,j], c]   (idx over Nin rows)
// ---------------------------------------------------------------------------
__global__ void pool_max(const float* __restrict__ feat, const int* __restrict__ idx,
                         float* __restrict__ out, int B, int Nin, int Nout, int C,
                         int ldin, int k, int ldout) {
    int t = blockIdx.x * blockDim.x + threadIdx.x;
    if (t >= B * Nout * C) return;
    int c = t % C;
    int i = (t / C) % Nout;
    int b = t / (C * Nout);
    float m = NEGBIG;
    for (int j = 0; j < k; ++j) {
        int id = idx[((size_t)b * Nout + i) * k + j];
        m = fmaxf(m, feat[((size_t)b * Nin + id) * ldin + c]);
    }
    out[((size_t)b * Nout + i) * ldout + c] = m;
}

// nearest source index (3D), strict < tie-break to lowest index (argmin)
__global__ void nearest_kernel(const float* __restrict__ tgt, const float* __restrict__ src,
                               int* __restrict__ out, int B, int Nt, int Ns, int tb, int sb) {
    int t = blockIdx.x * blockDim.x + threadIdx.x;
    if (t >= B * Nt) return;
    int b = t / Nt, n = t % Nt;
    const float* q = tgt + (size_t)b * tb + (size_t)n * 3;
    const float* s0 = src + (size_t)b * sb;
    float best = 3.4e38f; int bi = 0;
    for (int m = 0; m < Ns; ++m) {
        float dx = s0[m * 3] - q[0], dy = s0[m * 3 + 1] - q[1], dz = s0[m * 3 + 2] - q[2];
        float d2 = dx * dx + dy * dy + dz * dz;
        if (d2 < best) { best = d2; bi = m; }
    }
    out[t] = bi;
}

// final output: (B,2048,1664) = [fm5[nn2] (512) | feat1[nn1] (384) | feat2[nn1] (768)]
__global__ void final_gather(const float* __restrict__ fm5, const float* __restrict__ feat1,
                             const float* __restrict__ feat2, const int* __restrict__ nn1,
                             const int* __restrict__ nn2, float* __restrict__ out, int B, int N) {
    const int C = 1664;
    int t = blockIdx.x * blockDim.x + threadIdx.x;
    if (t >= B * N * C) return;
    int c = t % C;
    int n = (t / C) % N;
    int b = t / (C * N);
    float v;
    if (c < 512) {
        int j = nn2[(size_t)b * N + n];
        v = fm5[((size_t)b * 128 + j) * 512 + c];
    } else if (c < 896) {
        int j = nn1[(size_t)b * N + n];
        v = feat1[((size_t)b * 2048 + j) * 384 + (c - 512)];
    } else {
        int j = nn1[(size_t)b * N + n];
        v = feat2[((size_t)b * 512 + j) * 768 + (c - 896)];
    }
    out[t] = v;
}

// ---------------------------------------------------------------------------
// Host orchestration
// ---------------------------------------------------------------------------
static inline int cdiv(long long a, int b) { return (int)((a + b - 1) / b); }

extern "C" void kernel_launch(void* const* d_in, const int* in_sizes, int n_in,
                              void* d_out, int out_size, void* d_ws, size_t ws_size,
                              hipStream_t stream) {
    (void)in_sizes; (void)n_in; (void)out_size; (void)ws_size;
    const int B = 2, N1 = 2048, N2 = 512, N3 = 128, K = 10, KP = 4;

    const float* vertices = (const float*)d_in[0];
    const float* xyz      = (const float*)d_in[1];
    const float* normal   = (const float*)d_in[2];
    const float* brpts[3] = {vertices, xyz, normal};
    auto P = [&](int i) { return (const float*)d_in[3 + i]; };
    // per-branch param block: d0, w1,b1,d1,g1,be1, w2,b2,d2,g2,be2, w3,b3,d3,g3,be3

    // ---------------- workspace layout (256B-aligned slabs) ----------------
    float* ws = (float*)d_ws;
    size_t off = 0;
    auto alloc = [&](size_t nfl) { float* p = ws + off; off += (nfl + 63) & ~(size_t)63; return p; };
    int*   ni      = (int*)alloc((size_t)B * N1 * K);
    int*   ni2     = (int*)alloc((size_t)B * N2 * K);
    int*   ni3     = (int*)alloc((size_t)B * N3 * K);
    int*   idx4    = (int*)alloc((size_t)B * N2 * KP);
    int*   nn1     = (int*)alloc((size_t)B * N1);
    int*   nn2     = (int*)alloc((size_t)B * N1);
    float* featF   = alloc((size_t)B * N1 * 9);
    float* feat1   = alloc((size_t)B * N1 * 384);
    float* feat2   = alloc((size_t)B * N2 * 768);
    float* f0      = alloc((size_t)B * N1 * 128);
    float* fp      = alloc((size_t)B * N2 * 128);
    float* f2buf   = alloc((size_t)B * N2 * 256);
    float* fmp2    = alloc((size_t)B * N3 * 768);
    float* fm4     = alloc((size_t)B * N3 * 256);
    float* fm5     = alloc((size_t)B * N3 * 512);
    float* supn    = alloc((size_t)9 * 3584);
    float* dirnbuf = alloc((size_t)B * N1 * K * 3);      // >= all dirn uses
    float* meanb   = alloc(768);
    float* varb    = alloc(768);
    bf16_t* Abf    = (bf16_t*)alloc((size_t)4096 * 128 / 2);
    bf16_t* Wbf    = (bf16_t*)alloc((size_t)2048 * 768 / 2);
    float* ffull   = alloc((size_t)4096 * 1024);

    auto gemm = [&](const float* Af, const float* Wf, const float* bias, float* Cout,
                    int M, int Nc, int Kc) {
        int nA = M * Kc;
        cvt_bf16<<<cdiv(nA, 256), 256, 0, stream>>>(Af, Abf, nA);
        int nW = Kc * Nc;
        transpose_cvt<<<cdiv(nW, 256), 256, 0, stream>>>(Wf, Wbf, Kc, Nc);
        int blocks = (M / 128) * (Nc / 16);
        size_t shmem = (size_t)16 * (Kc + 8) * sizeof(bf16_t);
        wmma_gemm_bias<<<blocks, 256, shmem, stream>>>(Abf, Wbf, bias, Cout, M, Nc, Kc);
    };
    auto bn = [&](float* x, const float* g, const float* be, int rows, int ld, int C) {
        bn_stats<<<C, 256, 0, stream>>>(x, meanb, varb, rows, ld);
        bn_apply_relu<<<cdiv((long long)rows * C, 256), 256, 0, stream>>>(x, meanb, varb, g, be, rows, ld, C);
    };

    // ---------------- level 0 prep ----------------
    build_featF<<<cdiv((long long)B * N1 * 9, 256), 256, 0, stream>>>(vertices, xyz, normal, featF, B * N1);
    knn_kernel<<<cdiv(B * N1, 128), 128, 0, stream>>>(vertices, ni, B, N1, N1, 3, N1 * 3, K);

    // ---------------- level 1 (per branch): conv_surface -> conv_layer -> bn ----------------
    for (int bi = 0; bi < 3; ++bi) {
        const float* pts = brpts[bi];
        int pb = 16 * bi;
        dirn_kernel<<<cdiv(B * N1 * K, 256), 256, 0, stream>>>(pts, ni, dirnbuf, B, N1, K, 3, N1 * 3);
        norm_cols<<<cdiv(896, 128), 128, 0, stream>>>(P(pb + 0), supn, 3, 896);
        conv_surface_kernel<<<B * N1, 128, 0, stream>>>(dirnbuf, supn, f0, K);
        // f1 = conv_layer(ni, pts, f0, w1, b1, d1, 128)
        norm_cols<<<cdiv(896, 128), 128, 0, stream>>>(P(pb + 3), supn, 3, 896);
        gemm(f0, P(pb + 1), P(pb + 2), ffull, B * N1, 1024, 128);
        conv_layer_kernel<<<B * N1, 128, 0, stream>>>(dirnbuf, supn, ni, ffull,
                                                      feat1 + bi * 128, N1, K, 3, 128, 1024, 384);
        bn(feat1 + bi * 128, P(pb + 4), P(pb + 5), B * N1, 384, 128);
    }

    // ---------------- level 2 ----------------
    // ni2 over vertices[:, :512]
    knn_kernel<<<cdiv(B * N2, 128), 128, 0, stream>>>(vertices, ni2, B, N2, N2, 3, N1 * 3, K);
    for (int bi = 0; bi < 3; ++bi) {
        const float* pts = brpts[bi];
        int pb = 16 * bi;
        // pool branch features: knn(pts,4) on full N1, keep first N2 queries
        knn_kernel<<<cdiv(B * N2, 128), 128, 0, stream>>>(pts, idx4, B, N1, N2, 3, N1 * 3, KP);
        pool_max<<<cdiv((long long)B * N2 * 128, 256), 256, 0, stream>>>(
            feat1 + bi * 128, idx4, fp, B, N1, N2, 128, 384, KP, 128);
        dirn_kernel<<<cdiv(B * N2 * K, 256), 256, 0, stream>>>(pts, ni2, dirnbuf, B, N2, K, 3, N1 * 3);
        // f2 = bn_relu(conv_layer(ni2, vp, fp, w2, b2, d2, 256))
        norm_cols<<<cdiv(1792, 128), 128, 0, stream>>>(P(pb + 8), supn, 3, 1792);
        gemm(fp, P(pb + 6), P(pb + 7), ffull, B * N2, 2048, 128);
        conv_layer_kernel<<<B * N2, 256, 0, stream>>>(dirnbuf, supn, ni2, ffull,
                                                      f2buf, N2, K, 3, 256, 2048, 256);
        bn(f2buf, P(pb + 9), P(pb + 10), B * N2, 256, 256);
        // fm3 = bn_relu(conv_layer(ni2, vp, f2, w3, b3, d3, 256)) -> feat2 slice
        norm_cols<<<cdiv(1792, 128), 128, 0, stream>>>(P(pb + 13), supn, 3, 1792);
        gemm(f2buf, P(pb + 11), P(pb + 12), ffull, B * N2, 2048, 256);
        conv_layer_kernel<<<B * N2, 256, 0, stream>>>(dirnbuf, supn, ni2, ffull,
                                                      feat2 + bi * 256, N2, K, 3, 256, 2048, 768);
        bn(feat2 + bi * 256, P(pb + 14), P(pb + 15), B * N2, 768, 256);
    }

    // ---------------- level 3 ----------------
    // pool_2 = featF[:, :128]; fm_pool_2 = max-pool of feat_2 over knn(pool_1=featF[:, :512], 4)
    knn_kernel<<<cdiv(B * N3, 128), 128, 0, stream>>>(featF, idx4, B, N2, N3, 9, N1 * 9, KP);
    pool_max<<<cdiv((long long)B * N3 * 768, 256), 256, 0, stream>>>(
        feat2, idx4, fmp2, B, N2, N3, 768, 768, KP, 768);
    knn_kernel<<<cdiv(B * N3, 128), 128, 0, stream>>>(featF, ni3, B, N3, N3, 9, N1 * 9, K);
    dirn_kernel<<<cdiv(B * N3 * K, 256), 256, 0, stream>>>(featF, ni3, dirnbuf, B, N3, K, 9, N1 * 9);
    // fm_4 = conv_layer(ni3, pool_2, fm_pool_2, w4, b4, d4, 256)
    norm_cols<<<cdiv(1792, 128), 128, 0, stream>>>(P(50), supn, 9, 1792);
    gemm(fmp2, P(48), P(49), ffull, B * N3, 2048, 768);
    conv_layer_kernel<<<B * N3, 256, 0, stream>>>(dirnbuf, supn, ni3, ffull,
                                                  fm4, N3, K, 9, 256, 2048, 256);
    // fm_5 = conv_layer(ni3, pool_2, fm_4, w5, b5, d5, 512)
    norm_cols<<<cdiv(3584, 128), 128, 0, stream>>>(P(53), supn, 9, 3584);
    gemm(fm4, P(51), P(52), ffull, B * N3, 4096, 256);
    conv_layer_kernel<<<B * N3, 512, 0, stream>>>(dirnbuf, supn, ni3, ffull,
                                                  fm5, N3, K, 9, 512, 4096, 512);

    // ---------------- upsample + output ----------------
    nearest_kernel<<<cdiv(B * N1, 128), 128, 0, stream>>>(vertices, vertices, nn1, B, N1, N2, N1 * 3, N1 * 3);
    nearest_kernel<<<cdiv(B * N1, 128), 128, 0, stream>>>(vertices, vertices, nn2, B, N1, N3, N1 * 3, N1 * 3);
    final_gather<<<cdiv((long long)B * N1 * 1664, 256), 256, 0, stream>>>(
        fm5, feat1, feat2, nn1, nn2, (float*)d_out, B, N1);
}
